// InputAttention_77214922047718
// MI455X (gfx1250) — matrix operand order repaired and checked by reading, not compiled, not run on hardware
//
#include <hip/hip_runtime.h>

// Problem constants (from reference): BS=2048, T=128, N=256, M=256
#define kBS 2048
#define kT  128
#define kN  256
#define kM  256

typedef __attribute__((ext_vector_type(16))) __bf16 v16bf;
typedef __attribute__((ext_vector_type(8)))  float  v8f;
typedef __attribute__((ext_vector_type(4)))  int    v4i;

__device__ __forceinline__ unsigned short f2bf(float f) {
  unsigned int u = __float_as_uint(f);
  u += 0x7FFFu + ((u >> 16) & 1u);           // round-to-nearest-even
  return (unsigned short)(u >> 16);
}
__device__ __forceinline__ float bf_lo(unsigned int u) {  // low bf16 of a dword
  return __uint_as_float(u << 16);
}
__device__ __forceinline__ float bf_hi(unsigned int u) {  // high bf16 of a dword
  return __uint_as_float(u & 0xffff0000u);
}
__device__ __forceinline__ float fast_tanh(float x) {
#if __has_builtin(__builtin_amdgcn_tanhf)
  return __builtin_amdgcn_tanhf(x);          // native V_TANH_F32 on gfx1250
#elif __has_builtin(__builtin_amdgcn_tanh_f32)
  return __builtin_amdgcn_tanh_f32(x);
#else
  return tanhf(x);
#endif
}
__device__ __forceinline__ float sigm(float x) { return 1.0f / (1.0f + __expf(-x)); }

// ---------------------------------------------------------------------------
// Async global->LDS 16B copy (CDNA5 GLOBAL_LOAD_ASYNC_TO_LDS_B128, ASYNCcnt).
// Per-lane LDS destination address comes from a VGPR -> supports our swizzled
// fragment placement directly, no VGPR round-trip.
// Probe-confirmed signature: (v4i addrspace(1)*, v4i addrspace(3)*, Ii, Ii)
// ---------------------------------------------------------------------------
#if __has_builtin(__builtin_amdgcn_global_load_async_to_lds_b128)
#define ASYNC_BUILTIN 1
#else
#define ASYNC_BUILTIN 0
#endif

__device__ __forceinline__ void stage16(unsigned short* lds_dst,
                                        const unsigned short* gsrc) {
#if ASYNC_BUILTIN
  __builtin_amdgcn_global_load_async_to_lds_b128(
      (__attribute__((address_space(1))) v4i*)gsrc,
      (__attribute__((address_space(3))) v4i*)lds_dst, 0, 0);
#else
  unsigned int ldsa = (unsigned int)(unsigned long long)(void*)lds_dst;
  asm volatile("global_load_async_to_lds_b128 %0, %1, off"
               :: "v"(ldsa), "v"(gsrc) : "memory");
#endif
}
__device__ __forceinline__ void stage_wait() {
#if ASYNC_BUILTIN && __has_builtin(__builtin_amdgcn_s_wait_asynccnt)
  __builtin_amdgcn_s_wait_asynccnt(0);
#else
  asm volatile("s_wait_asynccnt 0" ::: "memory");
#endif
}

// ---- LDS swizzle into WMMA bf16 fragments, k = 16h + s in both A and B (ushort units) ----
// A (16x32, MxK): lane = ((k>>4)&1)*16 + (m&15); elem = k&15
__device__ __forceinline__ int ldsA_idx(int m, int k) {
  int lane = (((k >> 4) & 1) << 4) | (m & 15);
  return ((m >> 4) << 9) + (lane << 4) + (k & 15);
}
// B (32x16, KxN): lane = ((k>>4)&1)*16 + (n&15); elem = k&15
__device__ __forceinline__ int ldsB_idx(int k, int n) {
  int lane = (((k >> 4) & 1) << 4) | (n & 15);
  return ((n >> 4) << 9) + (lane << 4) + (k & 15);
}

// ---------------------------------------------------------------------------
// Tiled bf16 WMMA GEMM with double-buffered async-to-LDS staging.
// C(MxN) = A(MxK,row-major bf16) * B(KxN)
//   B_ROW_F32=true : B is row-major f32 (converted to bf16 while staging; sync)
//   B_ROW_F32=false: B memory holds W[n][k] ("column-major" bf16), ld=K (async)
//   C_BF16: store C as bf16, else f32
// 256 threads = 8 waves (2x4). Wave tile (16*FM)x(16*FN); block (32*FM)x(64*FN)
// ---------------------------------------------------------------------------
template <int FM, int FN, bool B_ROW_F32, bool C_BF16>
__global__ __launch_bounds__(256)
void gemm_bf16_wmma(const unsigned short* __restrict__ A, int lda, long long strideA,
                    const void* __restrict__ Bv, int ldb, long long strideB,
                    void* __restrict__ Cv, int ldc, long long strideC, int K)
{
  constexpr int BM = 32 * FM;
  constexpr int BN = 64 * FN;
  __shared__ __align__(32) unsigned short sA[2][BM * 32];
  __shared__ __align__(32) unsigned short sB[2][32 * BN];

  const int tid  = threadIdx.x;
  const int lane = tid & 31;
  const int wid  = tid >> 5;
  const int widm = wid >> 2;
  const int widn = wid & 3;
  const long long m0 = (long long)blockIdx.x * BM;
  const long long n0 = (long long)blockIdx.y * BN;
  const int batch = blockIdx.z;

  const unsigned short* Ab = A + (long long)batch * strideA;

  auto stageA = [&](int buf, int kc) {
    for (int g = tid; g < BM * 4; g += 256) {
      const int m  = g >> 2;
      const int k8 = (g & 3) << 3;
      stage16(&sA[buf][ldsA_idx(m, k8)], Ab + (m0 + m) * lda + (kc + k8));
    }
  };
  auto stageB = [&](int buf, int kc) {
    if (B_ROW_F32) {
      const float* Bb = (const float*)Bv + (long long)batch * strideB;
      for (int g = tid; g < 4 * BN; g += 256) {
        const int k  = g / (BN / 8);
        const int n8 = (g % (BN / 8)) << 3;
        const float* src = Bb + (long long)(kc + k) * ldb + n0 + n8;
        float4 v0 = ((const float4*)src)[0];
        float4 v1 = ((const float4*)src)[1];
        float vals[8] = {v0.x, v0.y, v0.z, v0.w, v1.x, v1.y, v1.z, v1.w};
#pragma unroll
        for (int i = 0; i < 8; ++i) sB[buf][ldsB_idx(k, n8 + i)] = f2bf(vals[i]);
      }
    } else {
      const unsigned short* Bb = (const unsigned short*)Bv + (long long)batch * strideB;
      for (int g = tid; g < 4 * BN; g += 256) {
        const int n  = g >> 2;
        const int k8 = (g & 3) << 3;
        stage16(&sB[buf][ldsB_idx(k8, n)], Bb + (n0 + n) * ldb + (kc + k8));
      }
    }
  };

  v8f acc[FM][FN];
#pragma unroll
  for (int i = 0; i < FM; ++i)
#pragma unroll
    for (int j = 0; j < FN; ++j)
#pragma unroll
      for (int r = 0; r < 8; ++r) acc[i][j][r] = 0.0f;

  stageA(0, 0); stageB(0, 0);
  stage_wait();
  __syncthreads();

  for (int kc = 0; kc < K; kc += 32) {
    const int cur = (kc >> 5) & 1, nxt = cur ^ 1;
    if (kc + 32 < K) { stageA(nxt, kc + 32); stageB(nxt, kc + 32); }

    v16bf afrag[FM];
#pragma unroll
    for (int mi = 0; mi < FM; ++mi)
      afrag[mi] = *(const v16bf*)(&sA[cur][((widm * FM + mi) << 9) + (lane << 4)]);
#pragma unroll
    for (int ni = 0; ni < FN; ++ni) {
      v16bf bfrag = *(const v16bf*)(&sB[cur][((widn * FN + ni) << 9) + (lane << 4)]);
#pragma unroll
      for (int mi = 0; mi < FM; ++mi)
        acc[mi][ni] = __builtin_amdgcn_wmma_f32_16x16x32_bf16(
            false, afrag[mi], false, bfrag, (short)0, acc[mi][ni], false, false);
    }
    stage_wait();
    __syncthreads();
  }

  const int nn  = lane & 15;
  const int grp = lane >> 4;
#pragma unroll
  for (int mi = 0; mi < FM; ++mi) {
#pragma unroll
    for (int ni = 0; ni < FN; ++ni) {
      const long long mbase = m0 + widm * (16 * FM) + mi * 16 + grp * 8;
      const long long n     = n0 + widn * (16 * FN) + ni * 16 + nn;
#pragma unroll
      for (int r = 0; r < 8; ++r) {
        const long long off = (long long)batch * strideC + (mbase + r) * ldc + n;
        if (C_BF16) ((unsigned short*)Cv)[off] = f2bf(acc[mi][ni][r]);
        else        ((float*)Cv)[off]          = acc[mi][ni][r];
      }
    }
  }
}

// ---------------------------------------------------------------------------
// Fused gates-GEMM + LSTM cell. Weights are row-PERMUTED: column j' = 4*u + g
// (g: 0=i,1=f,2=g,3=o), so a 128-wide tile holds 32 complete LSTM units.
// GEMM: G(2048x1024) = xh(2048x512) @ Wcat_p^T ; then per (b,u):
//   c' = sig(f)*c + sig(i)*tanh(g); h = sig(o)*tanh(c')
// Writes out[b,u,t], c, hs=[h|c] (bf16), and next-step xh_nx[:,256:512]=h.
// ---------------------------------------------------------------------------
__global__ __launch_bounds__(256)
void gates_lstm_wmma(const unsigned short* __restrict__ A /*xh_in*/,
                     const unsigned short* __restrict__ Bw /*Wcat_p, 1024x512*/,
                     const float* __restrict__ bsum_p,
                     float* __restrict__ c, float* __restrict__ out,
                     unsigned short* __restrict__ hs,
                     unsigned short* __restrict__ xh_nx, int t)
{
  constexpr int FM = 2, FN = 2, BM = 64, BN = 128, K = 512;
  __shared__ __align__(32) unsigned short sA[2][BM * 32];
  __shared__ __align__(32) unsigned short sB[2][32 * BN];
  __shared__ __align__(16) float sG[BM * BN];   // 32 KB gate staging

  const int tid  = threadIdx.x;
  const int lane = tid & 31;
  const int wid  = tid >> 5;
  const int widm = wid >> 2;
  const int widn = wid & 3;
  const long long m0 = (long long)blockIdx.x * BM;   // batch tile
  const long long n0 = (long long)blockIdx.y * BN;   // permuted-gate tile

  auto stageA = [&](int buf, int kc) {
    for (int g = tid; g < BM * 4; g += 256) {
      const int m  = g >> 2;
      const int k8 = (g & 3) << 3;
      stage16(&sA[buf][ldsA_idx(m, k8)], A + (m0 + m) * 512 + (kc + k8));
    }
  };
  auto stageB = [&](int buf, int kc) {
    for (int g = tid; g < 4 * BN; g += 256) {
      const int n  = g >> 2;
      const int k8 = (g & 3) << 3;
      stage16(&sB[buf][ldsB_idx(k8, n)], Bw + (n0 + n) * 512 + (kc + k8));
    }
  };

  v8f acc[FM][FN];
#pragma unroll
  for (int i = 0; i < FM; ++i)
#pragma unroll
    for (int j = 0; j < FN; ++j)
#pragma unroll
      for (int r = 0; r < 8; ++r) acc[i][j][r] = 0.0f;

  stageA(0, 0); stageB(0, 0);
  stage_wait();
  __syncthreads();

  for (int kc = 0; kc < K; kc += 32) {
    const int cur = (kc >> 5) & 1, nxt = cur ^ 1;
    if (kc + 32 < K) { stageA(nxt, kc + 32); stageB(nxt, kc + 32); }

    v16bf afrag[FM];
#pragma unroll
    for (int mi = 0; mi < FM; ++mi)
      afrag[mi] = *(const v16bf*)(&sA[cur][((widm * FM + mi) << 9) + (lane << 4)]);
#pragma unroll
    for (int ni = 0; ni < FN; ++ni) {
      v16bf bfrag = *(const v16bf*)(&sB[cur][((widn * FN + ni) << 9) + (lane << 4)]);
#pragma unroll
      for (int mi = 0; mi < FM; ++mi)
        acc[mi][ni] = __builtin_amdgcn_wmma_f32_16x16x32_bf16(
            false, afrag[mi], false, bfrag, (short)0, acc[mi][ni], false, false);
    }
    stage_wait();
    __syncthreads();
  }

  // stage accumulators into LDS so each thread can gather 4 gates of one unit
  const int nn  = lane & 15;
  const int grp = lane >> 4;
#pragma unroll
  for (int mi = 0; mi < FM; ++mi) {
#pragma unroll
    for (int ni = 0; ni < FN; ++ni) {
      const int lm = widm * 32 + mi * 16 + grp * 8;
      const int lc = widn * 32 + ni * 16 + nn;
#pragma unroll
      for (int r = 0; r < 8; ++r) sG[(lm + r) * BN + lc] = acc[mi][ni][r];
    }
  }
  __syncthreads();

  // LSTM pointwise: 64 batches x 32 units per block, 8 per thread
  for (int q = tid; q < BM * (BN / 4); q += 256) {
    const int bi = q >> 5;            // 0..63
    const int ui = q & 31;            // 0..31
    const long long b = m0 + bi;
    const long long u = (n0 >> 2) + ui;
    float4 gv = *(const float4*)(&sG[bi * BN + ui * 4]);
    float4 bb = *(const float4*)(&bsum_p[n0 + 4 * ui]);
    float gi = gv.x + bb.x, gf = gv.y + bb.y, gg = gv.z + bb.z, go = gv.w + bb.w;
    float cold = c[b * 256 + u];
    float cn = sigm(gf) * cold + sigm(gi) * fast_tanh(gg);
    float h  = sigm(go) * fast_tanh(cn);
    c[b * 256 + u] = cn;
    out[(b * 256 + u) * 128 + t] = h;              // out (bs, m, T)
    hs[b * 512 + u]        = f2bf(h);
    hs[b * 512 + 256 + u]  = f2bf(cn);
    xh_nx[b * 512 + 256 + u] = f2bf(h);            // h operand for NEXT step
  }
}

// ---------------------------------------------------------------------------
// One-time conversion. Wcat_p is gate-interleaved: row j' = 4*u + g maps to
// original row j = g*256 + u of [W_ih | W_hh]; bsum_p likewise.
// ---------------------------------------------------------------------------
__global__ __launch_bounds__(256)
void convert_kernel(const float* __restrict__ W_e, const float* __restrict__ U_e,
                    const float* __restrict__ W_ih, const float* __restrict__ W_hh,
                    const float* __restrict__ b_ih, const float* __restrict__ b_hh,
                    unsigned short* __restrict__ We_bf, unsigned short* __restrict__ Ue_bf,
                    unsigned short* __restrict__ Wcat_p, float* __restrict__ bsum_p)
{
  int idx = blockIdx.x * 256 + threadIdx.x;
  if (idx < 1024 * 512) {
    int jp = idx >> 9, k = idx & 511;
    int u = jp >> 2, g = jp & 3;
    int j = g * 256 + u;
    float w = (k < 256) ? W_ih[j * 256 + k] : W_hh[j * 256 + (k - 256)];
    Wcat_p[idx] = f2bf(w);
  }
  if (idx < 128 * 512) We_bf[idx] = f2bf(W_e[idx]);
  if (idx < 128 * 128) Ue_bf[idx] = f2bf(U_e[idx]);
  if (idx < 1024) {
    int u = idx >> 2, g = idx & 3;
    int j = g * 256 + u;
    bsum_p[idx] = b_ih[j] + b_hh[j];
  }
}

// Zero state (ws is poisoned by the harness -> init every call)
__global__ __launch_bounds__(256)
void init_state(unsigned short* __restrict__ hs, unsigned short* __restrict__ xhA,
                unsigned short* __restrict__ xhB, float* __restrict__ c)
{
  int idx = blockIdx.x * 256 + threadIdx.x;   // grid covers 2048*512
  hs[idx] = 0; xhA[idx] = 0; xhB[idx] = 0;
  if (idx < kBS * 256) c[idx] = 0.0f;
}

// ---------------------------------------------------------------------------
// Per-step attention, wave-per-batch (8 waves/block):
//   e[b,n] = sum_t V[t]*tanh(Ux[b,t,n] + whs[b,t])   (b128 loads: 8 cols/lane)
//   softmax over n=256 via wave32 shfl-xor butterflies; xt=alpha*e -> xh bf16
// ---------------------------------------------------------------------------
__global__ __launch_bounds__(256)
void attention_kernel(const unsigned short* __restrict__ Ux, const float* __restrict__ whs,
                      const float* __restrict__ Ve, unsigned short* __restrict__ xh)
{
  __shared__ float s_v[128];
  __shared__ float s_w[8][128];
  const int tid  = threadIdx.x;
  const int wv   = tid >> 5;
  const int lane = tid & 31;
  const long long b0 = (long long)blockIdx.x * 8;

  if (tid < 128) s_v[tid] = Ve[tid];
  for (int i = tid; i < 1024; i += 256) {
    int bq = i >> 7, tt = i & 127;
    s_w[bq][tt] = whs[(b0 + bq) * 128 + tt];
  }
  __syncthreads();

  const long long b = b0 + wv;
  const unsigned short* p = Ux + b * (kT * kN) + lane * 8;

  float e[8];
#pragma unroll
  for (int j = 0; j < 8; ++j) e[j] = 0.0f;

#pragma unroll 2
  for (int t = 0; t < kT; ++t) {
    uint4 raw = *(const uint4*)(p + (long long)t * kN);
    const float wt = s_w[wv][t];
    const float vt = s_v[t];
    e[0] += vt * fast_tanh(bf_lo(raw.x) + wt);
    e[1] += vt * fast_tanh(bf_hi(raw.x) + wt);
    e[2] += vt * fast_tanh(bf_lo(raw.y) + wt);
    e[3] += vt * fast_tanh(bf_hi(raw.y) + wt);
    e[4] += vt * fast_tanh(bf_lo(raw.z) + wt);
    e[5] += vt * fast_tanh(bf_hi(raw.z) + wt);
    e[6] += vt * fast_tanh(bf_lo(raw.w) + wt);
    e[7] += vt * fast_tanh(bf_hi(raw.w) + wt);
  }

  // wave-wide softmax over 256 values (8 per lane)
  float mx = e[0];
#pragma unroll
  for (int j = 1; j < 8; ++j) mx = fmaxf(mx, e[j]);
#pragma unroll
  for (int off = 16; off > 0; off >>= 1) mx = fmaxf(mx, __shfl_xor(mx, off, 32));

  float ex[8], sum = 0.0f;
#pragma unroll
  for (int j = 0; j < 8; ++j) { ex[j] = __expf(e[j] - mx); sum += ex[j]; }
#pragma unroll
  for (int off = 16; off > 0; off >>= 1) sum += __shfl_xor(sum, off, 32);
  const float inv = 1.0f / sum;

  unsigned int o0 = (unsigned int)f2bf(ex[0] * inv * e[0]) |
                    ((unsigned int)f2bf(ex[1] * inv * e[1]) << 16);
  unsigned int o1 = (unsigned int)f2bf(ex[2] * inv * e[2]) |
                    ((unsigned int)f2bf(ex[3] * inv * e[3]) << 16);
  unsigned int o2 = (unsigned int)f2bf(ex[4] * inv * e[4]) |
                    ((unsigned int)f2bf(ex[5] * inv * e[5]) << 16);
  unsigned int o3 = (unsigned int)f2bf(ex[6] * inv * e[6]) |
                    ((unsigned int)f2bf(ex[7] * inv * e[7]) << 16);
  uint4 st; st.x = o0; st.y = o1; st.z = o2; st.w = o3;
  *(uint4*)(xh + b * 512 + lane * 8) = st;          // xt = alpha * e (faithful)
}

extern "C" void kernel_launch(void* const* d_in, const int* in_sizes, int n_in,
                              void* d_out, int out_size, void* d_ws, size_t ws_size,
                              hipStream_t stream)
{
  (void)in_sizes; (void)n_in; (void)out_size; (void)ws_size;
  const float* x    = (const float*)d_in[0];
  const float* W_e  = (const float*)d_in[1];
  const float* U_e  = (const float*)d_in[2];
  const float* V_e  = (const float*)d_in[3];
  const float* W_ih = (const float*)d_in[4];
  const float* W_hh = (const float*)d_in[5];
  const float* b_ih = (const float*)d_in[6];
  const float* b_hh = (const float*)d_in[7];
  float* out = (float*)d_out;

  char* p = (char*)d_ws;
  auto take = [&](size_t bytes) { char* r = p; p += (bytes + 255) & ~(size_t)255; return r; };
  unsigned short* Ux_bf   = (unsigned short*)take((size_t)kBS * kT * kN * 2); // 128 MiB
  unsigned short* Wcat_p  = (unsigned short*)take((size_t)1024 * 512 * 2);
  unsigned short* We_bf   = (unsigned short*)take((size_t)128 * 512 * 2);
  unsigned short* Ue_bf   = (unsigned short*)take((size_t)128 * 128 * 2);
  float*          bsum_p  = (float*)take(1024 * 4);
  unsigned short* hs_bf   = (unsigned short*)take((size_t)kBS * 512 * 2);    // [h|c]
  unsigned short* xhA     = (unsigned short*)take((size_t)kBS * 512 * 2);    // [xt|h] ping
  unsigned short* xhB     = (unsigned short*)take((size_t)kBS * 512 * 2);    // [xt|h] pong
  float*          c_f32   = (float*)take((size_t)kBS * 256 * 4);
  float*          whs     = (float*)take((size_t)kBS * 128 * 4);

  convert_kernel<<<2048, 256, 0, stream>>>(W_e, U_e, W_ih, W_hh, b_ih, b_hh,
                                           We_bf, Ue_bf, Wcat_p, bsum_p);
  init_state<<<(kBS * 512) / 256, 256, 0, stream>>>(hs_bf, xhA, xhB, c_f32);

  // Ux[b] (T x N) = U_e (T x T) @ x[b] (T x N), stored bf16 as (b, t, n)
  gemm_bf16_wmma<4, 2, true, true>
      <<<dim3(1, kN / 128, kBS), 256, 0, stream>>>(
          Ue_bf, 128, 0, (const void*)x, kN, (long long)kT * kN,
          (void*)Ux_bf, kN, (long long)kT * kN, 128);

  for (int t = 0; t < kT; ++t) {
    unsigned short* xh_in = (t & 1) ? xhB : xhA;
    unsigned short* xh_nx = (t & 1) ? xhA : xhB;

    // whs (2048x128) = hs (2048x512) @ W_e^T
    gemm_bf16_wmma<1, 2, false, false>
        <<<dim3(kBS / 32, 1, 1), 256, 0, stream>>>(
            hs_bf, 512, 0, (const void*)We_bf, 512, 0, (void*)whs, 128, 0, 512);

    attention_kernel<<<kBS / 8, 256, 0, stream>>>(Ux_bf, whs, V_e, xh_in);

    gates_lstm_wmma<<<dim3(kBS / 64, 1024 / 128, 1), 256, 0, stream>>>(
        xh_in, Wcat_p, bsum_p, c_f32, out, hs_bf, xh_nx, t);
  }
}
